// Network_8512625181359
// MI455X (gfx1250) — matrix operand order repaired and checked by reading, not compiled, not run on hardware
//
#include <hip/hip_runtime.h>
#include <cstdint>

// ---------------------------------------------------------------------------
// Types for CDNA5 WMMA (wave32)
// ---------------------------------------------------------------------------
typedef __attribute__((ext_vector_type(16))) __bf16 v16bf;
typedef __attribute__((ext_vector_type(8)))  __bf16 v8bf;
typedef __attribute__((ext_vector_type(8)))  float  v8f;

static __device__ __forceinline__ unsigned short f32_to_bf16_bits(float f) {
    __bf16 h = (__bf16)f;
    return __builtin_bit_cast(unsigned short, h);
}

static __device__ __forceinline__ v16bf mk16(uint4 q0, uint4 q1) {
    return __builtin_shufflevector(__builtin_bit_cast(v8bf, q0),
                                   __builtin_bit_cast(v8bf, q1),
                                   0,1,2,3,4,5,6,7,8,9,10,11,12,13,14,15);
}

// ---------------------------------------------------------------------------
// f32 -> bf16 bits, identity layout (1x1 conv weights: K = C)
// ---------------------------------------------------------------------------
__global__ void k_f32_to_bf16(const float* __restrict__ in,
                              unsigned short* __restrict__ out, long n) {
    long i = blockIdx.x * (long)blockDim.x + threadIdx.x;
    if (i < n) out[i] = f32_to_bf16_bits(in[i]);
}

// ---------------------------------------------------------------------------
// f32 (O,C,3,3) -> bf16 bits, tap-major layout: out[m][r*C + c] = in[m][c][r]
// ---------------------------------------------------------------------------
__global__ void k_cvt_reorder9(const float* __restrict__ in,
                               unsigned short* __restrict__ out,
                               int C, long n /* O*C*9 */) {
    long i = blockIdx.x * (long)blockDim.x + threadIdx.x;
    if (i >= n) return;
    long C9 = (long)C * 9;
    long m   = i / C9;
    long rem = i - m * C9;
    int  r   = (int)(rem / C);
    int  c   = (int)(rem - (long)r * C);
    out[i] = f32_to_bf16_bits(in[m * C9 + (long)c * 9 + r]);
}

// ---------------------------------------------------------------------------
// Implicit-GEMM conv, single-wave variant (small O or C%32!=0).
//   Weights tap-major: W[m][k], k = r*C + c. 4 M-tiles x 16 pixels per wave.
// ---------------------------------------------------------------------------
__global__ void __launch_bounds__(32)
k_conv_wmma(const unsigned short* __restrict__ wts, long w_zstride,
            const float* __restrict__ in0, int c1,
            const float* __restrict__ in1, int c2,
            const float* __restrict__ bias,
            float* __restrict__ out,
            int O, int H, int W, int ksz) {
    const int lane = threadIdx.x;
    const int half = lane >> 4;
    const int l15  = lane & 15;
    const int HW   = H * W;
    const int N    = HW;
    const int C    = c1 + c2;
    const int K    = C * ksz * ksz;
    const int z    = blockIdx.z;
    const int m0   = blockIdx.y << 6;
    const int n0   = blockIdx.x << 4;
    const int pad  = ksz >> 1;

    const unsigned short* wbase = wts + (size_t)z * w_zstride;
    const float* i0 = in0 + (size_t)z * c1 * HW;
    const float* i1 = in1 ? (in1 + (size_t)z * c2 * HW) : (const float*)0;

    const unsigned short* wrow[4];
#pragma unroll
    for (int mt = 0; mt < 4; ++mt) {
        int m  = m0 + mt * 16 + l15;
        int mc = (m < O) ? m : (O - 1);
        wrow[mt] = wbase + (size_t)mc * K;
    }

    int n  = n0 + l15;
    int nc = (n < N) ? n : (N - 1);
    int py = nc / W, px = nc - py * W;

    v8f acc[4] = {};
    const int ka = half ? 16 : 0;

    if ((C & 31) == 0) {
        // fast path: each 32-wide K block sits inside one 3x3 tap
        int c0 = 0, kh = 0, kw = 0;
        for (int k0 = 0; k0 < K; k0 += 32) {
            int yy = py + kh - pad;
            int xx = px + kw - pad;
            bool inb = (n < N) && (yy >= 0) && (yy < H) && (xx >= 0) && (xx < W);
            long off = (long)yy * W + xx;

            int cb = c0 + (half ? 16 : 0);
            const float* basep = (cb < c1) ? (i0 + (size_t)cb * HW)
                                           : (i1 + (size_t)(cb - c1) * HW);
            v16bf b;
#pragma unroll
            for (int j = 0; j < 16; ++j) {
                float v = inb ? basep[(size_t)j * HW + off] : 0.0f;
                b[j] = (__bf16)v;
            }

#pragma unroll
            for (int mt = 0; mt < 4; ++mt) {
                uint4 q0 = *(const uint4*)(wrow[mt] + k0 + ka);
                uint4 q1 = *(const uint4*)(wrow[mt] + k0 + ka + 8);
                v16bf a = mk16(q0, q1);
                acc[mt] = __builtin_amdgcn_wmma_f32_16x16x32_bf16(
                              false, a, false, b, (short)0, acc[mt], false, false);
            }

            c0 += 32;
            if (c0 >= C) {
                c0 = 0;
                kw += 1;
                if (kw == ksz) { kw = 0; kh += 1; }
            }
        }
    } else {
        // generic path (tiny layers)
        for (int k0 = 0; k0 < K; k0 += 32) {
            v16bf b;
            int kb = k0 + (half ? 16 : 0);
#pragma unroll
            for (int j = 0; j < 16; ++j) {
                int k = kb + j;
                float v = 0.0f;
                if (k < K && n < N) {
                    int cch, rr;
                    if (ksz == 1) { cch = k; rr = 0; }
                    else { rr = k / C; cch = k - rr * C; }
                    int yy = py + (rr / 3) - pad;
                    int xx = px + (rr - (rr / 3) * 3) - pad;
                    if (yy >= 0 && yy < H && xx >= 0 && xx < W)
                        v = (cch < c1) ? i0[(size_t)cch * HW + yy * W + xx]
                                       : i1[(size_t)(cch - c1) * HW + yy * W + xx];
                }
                b[j] = (__bf16)v;
            }
#pragma unroll
            for (int mt = 0; mt < 4; ++mt) {
                v16bf a;
#pragma unroll
                for (int j = 0; j < 8; ++j) {
                    int k1 = k0 + ka + j;
                    int k2 = k0 + ka + 8 + j;
                    unsigned short w1 = (k1 < K) ? wrow[mt][k1] : (unsigned short)0;
                    unsigned short w2 = (k2 < K) ? wrow[mt][k2] : (unsigned short)0;
                    a[j]     = __builtin_bit_cast(__bf16, w1);
                    a[8 + j] = __builtin_bit_cast(__bf16, w2);
                }
                acc[mt] = __builtin_amdgcn_wmma_f32_16x16x32_bf16(
                              false, a, false, b, (short)0, acc[mt], false, false);
            }
        }
    }

    float* obase = out + (size_t)z * O * HW;
#pragma unroll
    for (int mt = 0; mt < 4; ++mt) {
#pragma unroll
        for (int vi = 0; vi < 8; ++vi) {
            int mm = m0 + mt * 16 + vi + half * 8;
            if (mm < O && n < N) {
                float rv = acc[mt][vi];
                if (bias) rv += bias[mm];
                obase[(size_t)mm * HW + n] = rv;
            }
        }
    }
}

// ---------------------------------------------------------------------------
// Implicit-GEMM conv, LDS-shared variant (O >= 256, C%32 == 0, shared wts).
//   4 waves per block cover 256 output channels and SHARE the 32x16 B tile:
//   cooperative gather -> LDS (double-buffered, 80B row pitch for bank
//   spread), each wave reads its fragment with 2x ds_load_b128, 4 WMMAs.
// ---------------------------------------------------------------------------
__global__ void __launch_bounds__(128)
k_conv_wmma_smem(const unsigned short* __restrict__ wts,
                 const float* __restrict__ in0, int c1,
                 const float* __restrict__ in1, int c2,
                 const float* __restrict__ bias,
                 float* __restrict__ out,
                 int O, int H, int W, int ksz) {
    __shared__ unsigned short lds[2 * 16 * 40];   // 2 buffers, 16 rows x 40 bf16

    const int tid  = threadIdx.x;
    const int lane = tid & 31;
    const int wave = tid >> 5;
    const int half = lane >> 4;
    const int l15  = lane & 15;
    const int HW   = H * W;
    const int N    = HW;
    const int C    = c1 + c2;
    const int K    = C * ksz * ksz;
    const int z    = blockIdx.z;
    const int m0   = (blockIdx.y * 4 + wave) << 6;   // 64 rows per wave
    const int n0   = blockIdx.x << 4;
    const int pad  = ksz >> 1;

    const float* i0 = in0 + (size_t)z * c1 * HW;
    const float* i1 = in1 ? (in1 + (size_t)z * c2 * HW) : (const float*)0;

    const unsigned short* wrow[4];
#pragma unroll
    for (int mt = 0; mt < 4; ++mt) {
        int m  = m0 + mt * 16 + l15;
        int mc = (m < O) ? m : (O - 1);
        wrow[mt] = wts + (size_t)mc * K;
    }

    // staging role: thread -> (pixel sp, 4 consecutive k at kb4)
    const int sp  = tid & 15;
    const int kb4 = (tid >> 4) << 2;
    int np   = n0 + sp;
    int npc  = (np < N) ? np : (N - 1);
    int spy  = npc / W, spx = npc - spy * W;
    bool pvalid = (np < N);

    int n = n0 + l15;   // this wave-lane's output pixel

    auto stage = [&](int buf, int c0, int kh, int kw) {
        int yy = spy + kh - pad;
        int xx = spx + kw - pad;
        bool inb = pvalid && (yy >= 0) && (yy < H) && (xx >= 0) && (xx < W);
        long off = (long)yy * W + xx;
        int cb = c0 + kb4;
        const float* basep = (cb < c1) ? (i0 + (size_t)cb * HW)
                                       : (i1 + (size_t)(cb - c1) * HW);
        float v0 = inb ? basep[off] : 0.0f;
        float v1 = inb ? basep[(size_t)HW + off] : 0.0f;
        float v2 = inb ? basep[2 * (size_t)HW + off] : 0.0f;
        float v3 = inb ? basep[3 * (size_t)HW + off] : 0.0f;
        uint2 pk;
        pk.x = (unsigned)f32_to_bf16_bits(v0) | ((unsigned)f32_to_bf16_bits(v1) << 16);
        pk.y = (unsigned)f32_to_bf16_bits(v2) | ((unsigned)f32_to_bf16_bits(v3) << 16);
        *(uint2*)&lds[buf * 640 + sp * 40 + kb4] = pk;
    };

    v8f acc[4] = {};
    const int ka = half ? 16 : 0;

    int c0 = 0, kh = 0, kw = 0;
    stage(0, c0, kh, kw);
    __syncthreads();

    int cur = 0;
    for (int k0 = 0; k0 < K; k0 += 32) {
        // read this wave's B fragment from LDS (16 consecutive k per lane)
        const unsigned short* rowp = &lds[cur * 640 + l15 * 40 + half * 16];
        uint4 q0 = *(const uint4*)(rowp);
        uint4 q1 = *(const uint4*)(rowp + 8);
        v16bf b = mk16(q0, q1);

        // stage next K block into the other buffer (overlaps with WMMAs)
        if (k0 + 32 < K) {
            c0 += 32;
            if (c0 >= C) {
                c0 = 0;
                kw += 1;
                if (kw == ksz) { kw = 0; kh += 1; }
            }
            stage(cur ^ 1, c0, kh, kw);
        }

#pragma unroll
        for (int mt = 0; mt < 4; ++mt) {
            uint4 a0 = *(const uint4*)(wrow[mt] + k0 + ka);
            uint4 a1 = *(const uint4*)(wrow[mt] + k0 + ka + 8);
            v16bf a = mk16(a0, a1);
            acc[mt] = __builtin_amdgcn_wmma_f32_16x16x32_bf16(
                          false, a, false, b, (short)0, acc[mt], false, false);
        }

        __syncthreads();
        cur ^= 1;
    }

    float* obase = out + (size_t)z * O * HW;
#pragma unroll
    for (int mt = 0; mt < 4; ++mt) {
#pragma unroll
        for (int vi = 0; vi < 8; ++vi) {
            int mm = m0 + mt * 16 + vi + half * 8;
            if (mm < O && n < N) {
                float rv = acc[mt][vi];
                if (bias) rv += bias[mm];
                obase[(size_t)mm * HW + n] = rv;
            }
        }
    }
}

// ---------------------------------------------------------------------------
// BatchNorm (training mode): per-channel mean/var over (B,H,W)
// ---------------------------------------------------------------------------
__global__ void k_bn_stats(const float* __restrict__ x, float* __restrict__ stats,
                           int C, int B, long hw) {
    int c = blockIdx.x;
    long n = (long)B * hw;
    float s = 0.f, s2 = 0.f;
    for (long i = threadIdx.x; i < n; i += blockDim.x) {
        long b = i / hw;
        long r = i - b * hw;
        float v = x[((long)b * C + c) * hw + r];
        s += v; s2 += v * v;
    }
    __shared__ float sh[256], sh2[256];
    sh[threadIdx.x] = s; sh2[threadIdx.x] = s2;
    __syncthreads();
    for (int st = blockDim.x >> 1; st > 0; st >>= 1) {
        if ((int)threadIdx.x < st) {
            sh[threadIdx.x]  += sh[threadIdx.x + st];
            sh2[threadIdx.x] += sh2[threadIdx.x + st];
        }
        __syncthreads();
    }
    if (threadIdx.x == 0) {
        float m = sh[0] / (float)n;
        stats[c]     = m;
        stats[C + c] = sh2[0] / (float)n - m * m;
    }
}

__global__ void k_bn_apply(float* __restrict__ x, const float* __restrict__ stats,
                           int C, long hw, long total, int relu) {
    long i = blockIdx.x * (long)blockDim.x + threadIdx.x;
    if (i >= total) return;
    int c = (int)((i / hw) % C);
    float y = (x[i] - stats[c]) * rsqrtf(stats[C + c] + 1e-5f);
    if (relu) y = fmaxf(y, 0.f);
    x[i] = y;
}

// ---------------------------------------------------------------------------
// ConvLSTM gate update (gates i,f,o,g)
// ---------------------------------------------------------------------------
__global__ void k_lstm(const float* __restrict__ g, const float* __restrict__ cprev,
                       float* __restrict__ cnew, float* __restrict__ h,
                       int hid, long hw, long total) {
    long i = blockIdx.x * (long)blockDim.x + threadIdx.x;
    if (i >= total) return;
    long bhw = (long)hid * hw;
    long bb  = i / bhw;
    long r   = i - bb * bhw;
    const float* gb = g + bb * 4 * bhw;
    float gi = gb[r];
    float gf = gb[bhw + r];
    float go = gb[2 * bhw + r];
    float gg = gb[3 * bhw + r];
    float si = 1.f / (1.f + __expf(-gi));
    float sf = 1.f / (1.f + __expf(-gf));
    float so = 1.f / (1.f + __expf(-go));
    float c  = sf * cprev[i] + si * tanhf(gg);
    cnew[i] = c;
    h[i]    = so * tanhf(c);
}

__global__ void k_maxpool2(const float* __restrict__ in, float* __restrict__ out,
                           int H, int W, long total) {
    long i = blockIdx.x * (long)blockDim.x + threadIdx.x;
    if (i >= total) return;
    int Wo = W >> 1, Ho = H >> 1;
    int x = (int)(i % Wo);
    long r = i / Wo;
    int y = (int)(r % Ho);
    long p = r / Ho;
    const float* ib = in + p * (long)H * W + (2 * y) * W + 2 * x;
    out[i] = fmaxf(fmaxf(ib[0], ib[1]), fmaxf(ib[W], ib[W + 1]));
}

// Bilinear 2x upsample (align_corners=True) fused with skip add.
__global__ void k_up2_add(const float* __restrict__ a, const float* __restrict__ addv,
                          float* __restrict__ out, int h, int w, long total) {
    long i = blockIdx.x * (long)blockDim.x + threadIdx.x;
    if (i >= total) return;
    int W2 = 2 * w, H2 = 2 * h;
    int x = (int)(i % W2);
    long r = i / W2;
    int y = (int)(r % H2);
    long bc = r / H2;
    float fy = (h > 1) ? ((float)(h - 1) / (float)(2 * h - 1)) : 0.f;
    float fx = (w > 1) ? ((float)(w - 1) / (float)(2 * w - 1)) : 0.f;
    float sy = y * fy, sx = x * fx;
    int y0 = (int)floorf(sy); int y1 = min(y0 + 1, h - 1); float wy = sy - y0;
    int x0 = (int)floorf(sx); int x1 = min(x0 + 1, w - 1); float wx = sx - x0;
    const float* ab = a + bc * (long)h * w;
    float top = ab[y0 * w + x0] * (1.f - wx) + ab[y0 * w + x1] * wx;
    float bot = ab[y1 * w + x0] * (1.f - wx) + ab[y1 * w + x1] * wx;
    out[i] = top * (1.f - wy) + bot * wy + addv[i];
}

__global__ void k_gap(const float* __restrict__ x, float* __restrict__ out, long hw) {
    float s = 0.f;
    const float* xb = x + (long)blockIdx.x * hw;
    for (long i = threadIdx.x; i < hw; i += blockDim.x) s += xb[i];
    __shared__ float sh[256];
    sh[threadIdx.x] = s;
    __syncthreads();
    for (int st = blockDim.x >> 1; st > 0; st >>= 1) {
        if ((int)threadIdx.x < st) sh[threadIdx.x] += sh[threadIdx.x + st];
        __syncthreads();
    }
    if (threadIdx.x == 0) out[blockIdx.x] = sh[0] / (float)hw;
}

__global__ void k_att(const float* __restrict__ gap, const float* __restrict__ attw,
                      float* __restrict__ att, int B, int Kk, int C) {
    int i = blockIdx.x * blockDim.x + threadIdx.x;
    if (i >= B * Kk) return;
    int b = i / Kk, k = i - b * Kk;
    float s = 0.f;
    for (int c = 0; c < C; ++c) s += gap[b * C + c] * attw[k * C + c];
    att[i] = 1.f / (1.f + __expf(-s));
}

// Per-sample CondConv weights, aggregated AND written in tap-major bf16 layout.
__global__ void k_wagg(const float* __restrict__ ccw, const float* __restrict__ att,
                       unsigned short* __restrict__ out,
                       int Kk, int O, int C, long total /* B*O*C*9 */) {
    long i = blockIdx.x * (long)blockDim.x + threadIdx.x;
    if (i >= total) return;
    long M = (long)O * C * 9;
    long b   = i / M;
    long rem = i - b * M;
    long C9  = (long)C * 9;
    long m    = rem / C9;
    long rem2 = rem - m * C9;
    int  r = (int)(rem2 / C);
    int  c = (int)(rem2 - (long)r * C);
    long src = (m * C + c) * 9 + r;
    float s = 0.f;
    for (int k = 0; k < Kk; ++k) s += att[b * Kk + k] * ccw[(long)k * M + src];
    out[i] = f32_to_bf16_bits(s);
}

__global__ void k_add_relu(const float* __restrict__ a, const float* __restrict__ b,
                           float* __restrict__ out, long n) {
    long i = blockIdx.x * (long)blockDim.x + threadIdx.x;
    if (i < n) out[i] = fmaxf(a[i] + b[i], 0.f);
}

__global__ void k_fill(float* __restrict__ p, long n, float v) {
    long i = blockIdx.x * (long)blockDim.x + threadIdx.x;
    if (i < n) p[i] = v;
}

__global__ void k_dis(const float* __restrict__ t, float* __restrict__ out, long n) {
    long i = blockIdx.x * (long)blockDim.x + threadIdx.x;
    if (i < n) out[i] = fabsf((t[n + i] - t[i]) * 0.5f);
}

// ---------------------------------------------------------------------------
// Host orchestration
// ---------------------------------------------------------------------------
extern "C" void kernel_launch(void* const* d_in, const int* in_sizes, int n_in,
                              void* d_out, int out_size, void* d_ws, size_t ws_size,
                              hipStream_t stream) {
    if (n_in < 50 || !d_out || !d_ws) return;
    const int B = 2;

    // ---- Input pointer mapping (handles both flattening conventions) ----
    const float *CLw[5][2], *CLb[5][2], *CONVW[5];
    struct Dec { const float *att, *ccw, *resw, *w1, *w2; } D[4];
    const float *HEADW, *HEADB, *P2VW, *P2VB, *T;
    auto P = [&](int i) { return (const float*)d_in[i]; };

    if (in_sizes[0] == 2 * 2 * 3 * 128 * 128) {
        int i = 0;
        T = P(i++); P2VW = P(i++); P2VB = P(i++);
        for (int l = 0; l < 5; ++l) {
            for (int c = 0; c < 2; ++c) { CLw[l][c] = P(i++); CLb[l][c] = P(i++); }
            CONVW[l] = P(i++);
        }
        for (int d = 0; d < 4; ++d) {
            D[d].w1 = P(i++); D[d].ccw = P(i++); D[d].att = P(i++);
            D[d].w2 = P(i++); D[d].resw = P(i++);
        }
        HEADW = P(i++); HEADB = P(i++);
    } else {
        int i = 0;
        for (int l = 0; l < 5; ++l)
            for (int c = 0; c < 2; ++c) { CLb[l][c] = P(i++); CLw[l][c] = P(i++); }
        for (int l = 0; l < 5; ++l) CONVW[l] = P(i++);
        for (int d = 0; d < 4; ++d) {
            D[d].att = P(i++); D[d].ccw = P(i++); D[d].resw = P(i++);
            D[d].w1 = P(i++); D[d].w2 = P(i++);
        }
        HEADB = P(i++); HEADW = P(i++); P2VB = P(i++); P2VW = P(i++); T = P(i++);
    }

    // ---- Workspace bump allocator ----
    size_t ws_off = 0;
    auto allocB = [&](size_t bytes) -> void* {
        size_t a = (ws_off + 255) & ~(size_t)255;
        ws_off = a + bytes;
        return (void*)((char*)d_ws + a);
    };
    auto allocF = [&](long n) -> float* { return (float*)allocB((size_t)n * 4); };
    auto allocU = [&](long n) -> unsigned short* { return (unsigned short*)allocB((size_t)n * 2); };

    unsigned short* wbuf = allocU(18874368);   // largest weight (cl4/cl5 cell)
    unsigned short* waggbuf = allocU(2 * 36864);
    float* statbuf = allocF(4096);
    float* gapbuf  = allocF(256);
    float* attbuf  = allocF(16);
    const long ZMAX = 2097152;                 // B*64*128*128
    float* zbuf = allocF(ZMAX);

    auto GB = [](long n) { return dim3((unsigned)((n + 255) / 256)); };

    // ---- Launch helpers ----
    auto cvtW1 = [&](const float* w, long n) {   // 1x1 weights: identity layout
        k_f32_to_bf16<<<GB(n), 256, 0, stream>>>(w, wbuf, n);
    };
    auto cvtW3 = [&](const float* w, int Ctot, long n) { // 3x3 weights: tap-major
        k_cvt_reorder9<<<GB(n), 256, 0, stream>>>(w, wbuf, Ctot, n);
    };
    auto convPre = [&](const unsigned short* wp, long wstride,
                       const float* in0, int c1, const float* in1, int c2,
                       const float* bias, float* out, int O, int H, int W, int ksz) {
        int C = c1 + c2;
        if (wstride == 0 && O >= 256 && (C & 31) == 0) {
            dim3 grid((unsigned)((H * W + 15) / 16), (unsigned)((O + 255) / 256), (unsigned)B);
            k_conv_wmma_smem<<<grid, 128, 0, stream>>>(wp, in0, c1, in1, c2, bias, out, O, H, W, ksz);
        } else {
            dim3 grid((unsigned)((H * W + 15) / 16), (unsigned)((O + 63) / 64), (unsigned)B);
            k_conv_wmma<<<grid, 32, 0, stream>>>(wp, wstride, in0, c1, in1, c2, bias, out, O, H, W, ksz);
        }
    };
    auto bnact = [&](float* x, int C, int H, int W, int relu) {
        long hw = (long)H * W;
        k_bn_stats<<<dim3((unsigned)C), 256, 0, stream>>>(x, statbuf, C, B, hw);
        long tot = (long)B * C * hw;
        k_bn_apply<<<GB(tot), 256, 0, stream>>>(x, statbuf, C, hw, tot, relu);
    };

    k_fill<<<GB(ZMAX), 256, 0, stream>>>(zbuf, ZMAX, 0.f);

    // ---- dis = |(t2-t1)*0.5| ; F = relu(bn(conv1x1(dis))) ----
    const long disN = (long)B * 3 * 128 * 128;
    float* dis = allocF(disN);
    k_dis<<<GB(disN), 256, 0, stream>>>(T, dis, disN);

    float* F = allocF((long)B * 64 * 128 * 128);
    cvtW1(P2VW, 64 * 3);
    convPre(wbuf, 0, dis, 3, nullptr, 0, P2VB, F, 64, 128, 128, 1);
    bnact(F, 64, 128, 128, 1);

    // Shared ConvLSTM scratch (sized for level 1, reused everywhere)
    float* gshared = allocF(4 * ZMAX);
    float* cshared = allocF(ZMAX);

    // ---- One ConvLSTM cell over T=2 timesteps ----
    auto runCell = [&](const float* x0, const float* x1, int cin, int hid,
                       int H, int W, const float* w, const float* bias,
                       float** s0_out, float** s1_out) {
        long hw  = (long)H * W;
        long hsz = (long)B * hid * hw;
        float* s0 = allocF(hsz);
        float* s1 = allocF(hsz);
        long nw = (long)(4 * hid) * (cin + hid) * 9;
        cvtW3(w, cin + hid, nw);
        // t=0: h_prev = 0, c_prev = 0
        convPre(wbuf, 0, x0, cin, zbuf, hid, bias, gshared, 4 * hid, H, W, 3);
        k_lstm<<<GB(hsz), 256, 0, stream>>>(gshared, zbuf, cshared, s0, hid, hw, hsz);
        // t=1
        convPre(wbuf, 0, x1, cin, s0, hid, bias, gshared, 4 * hid, H, W, 3);
        k_lstm<<<GB(hsz), 256, 0, stream>>>(gshared, cshared, cshared, s1, hid, hw, hsz);
        *s0_out = s0; *s1_out = s1;
    };
    auto runLevel = [&](const float* x0, const float* x1, int cin, int hid,
                        int H, int W, int lvl, float** v0, float** v1) {
        float *a0, *a1;
        runCell(x0, x1, cin, hid, H, W, CLw[lvl][0], CLb[lvl][0], &a0, &a1);
        runCell(a0, a1, hid, hid, H, W, CLw[lvl][1], CLb[lvl][1], v0, v1);
    };
    auto maxpool = [&](const float* in, float* out, int C, int H, int W) {
        long tot = (long)B * C * (H / 2) * (W / 2);
        k_maxpool2<<<GB(tot), 256, 0, stream>>>(in, out, H, W, tot);
    };
    auto ylayer = [&](const float* vlast, int C, int H, int W, int lvl) -> float* {
        float* y = allocF((long)B * C * H * W);
        cvtW3(CONVW[lvl], C, (long)C * C * 9);
        convPre(wbuf, 0, vlast, C, nullptr, 0, nullptr, y, C, H, W, 3);
        bnact(y, C, H, W, 1);
        return y;
    };

    // ---- Encoder: 5 ConvLSTM levels ----
    float *v0, *v1;
    runLevel(F, F, 64, 64, 128, 128, 0, &v0, &v1);
    float* y1 = ylayer(v1, 64, 128, 128, 0);

    float* d0 = allocF((long)B * 64 * 64 * 64);
    float* d1 = allocF((long)B * 64 * 64 * 64);
    maxpool(v0, d0, 64, 128, 128); maxpool(v1, d1, 64, 128, 128);
    runLevel(d0, d1, 64, 128, 64, 64, 1, &v0, &v1);
    float* y2 = ylayer(v1, 128, 64, 64, 1);

    d0 = allocF((long)B * 128 * 32 * 32); d1 = allocF((long)B * 128 * 32 * 32);
    maxpool(v0, d0, 128, 64, 64); maxpool(v1, d1, 128, 64, 64);
    runLevel(d0, d1, 128, 256, 32, 32, 2, &v0, &v1);
    float* y3 = ylayer(v1, 256, 32, 32, 2);

    d0 = allocF((long)B * 256 * 16 * 16); d1 = allocF((long)B * 256 * 16 * 16);
    maxpool(v0, d0, 256, 32, 32); maxpool(v1, d1, 256, 32, 32);
    runLevel(d0, d1, 256, 512, 16, 16, 3, &v0, &v1);
    float* y4 = ylayer(v1, 512, 16, 16, 3);

    d0 = allocF((long)B * 512 * 8 * 8); d1 = allocF((long)B * 512 * 8 * 8);
    maxpool(v0, d0, 512, 16, 16); maxpool(v1, d1, 512, 16, 16);
    runLevel(d0, d1, 512, 512, 8, 8, 4, &v0, &v1);
    float* y5 = ylayer(v1, 512, 8, 8, 4);

    // ---- Decoder: bottleneck(up2(x1)+x2) with CondConv ----
    auto decoder = [&](const float* x1, const float* x2, int cin, int cout,
                       int h, int w, const Dec& dp) -> float* {
        int H = 2 * h, W = 2 * w;
        int mid = cout / 4;
        long xin_n = (long)B * cin * H * W;
        float* xin = allocF(xin_n);
        k_up2_add<<<GB(xin_n), 256, 0, stream>>>(x1, x2, xin, h, w, xin_n);

        // residual branch
        float* res = allocF((long)B * cout * H * W);
        cvtW1(dp.resw, (long)cout * cin);
        convPre(wbuf, 0, xin, cin, nullptr, 0, nullptr, res, cout, H, W, 1);
        bnact(res, cout, H, W, 0);

        // main branch: 1x1 reduce
        float* ya = allocF((long)B * mid * H * W);
        cvtW1(dp.w1, (long)mid * cin);
        convPre(wbuf, 0, xin, cin, nullptr, 0, nullptr, ya, mid, H, W, 1);
        bnact(ya, mid, H, W, 1);

        // CondConv 3x3 (per-sample aggregated weights, tap-major bf16)
        long hw = (long)H * W;
        k_gap<<<dim3((unsigned)(B * mid)), 256, 0, stream>>>(ya, gapbuf, hw);
        k_att<<<1, 64, 0, stream>>>(gapbuf, dp.att, attbuf, B, 4, mid);
        long M = (long)mid * mid * 9;
        k_wagg<<<GB(B * M), 256, 0, stream>>>(dp.ccw, attbuf, waggbuf, 4, mid, mid, B * M);
        float* yb = allocF((long)B * mid * H * W);
        convPre(waggbuf, M, ya, mid, nullptr, 0, nullptr, yb, mid, H, W, 3);
        bnact(yb, mid, H, W, 1);

        // 1x1 expand
        float* yc = allocF((long)B * cout * H * W);
        cvtW1(dp.w2, (long)cout * mid);
        convPre(wbuf, 0, yb, mid, nullptr, 0, nullptr, yc, cout, H, W, 1);
        bnact(yc, cout, H, W, 0);

        long tot = (long)B * cout * H * W;
        k_add_relu<<<GB(tot), 256, 0, stream>>>(res, yc, res, tot);
        return res;
    };

    float* o = decoder(y5, y4, 512, 256, 8, 8, D[0]);
    o = decoder(o, y3, 256, 128, 16, 16, D[1]);
    o = decoder(o, y2, 128, 64, 32, 32, D[2]);
    o = decoder(o, y1, 64, 64, 64, 64, D[3]);

    // ---- Head: 3x3 conv, 2 output channels, + bias ----
    cvtW3(HEADW, 64, (long)2 * 64 * 9);
    convPre(wbuf, 0, o, 64, nullptr, 0, HEADB, (float*)d_out, 2, 128, 128, 3);

    (void)in_sizes; (void)n_in; (void)out_size; (void)ws_size;
}
